// ScaledDotProductAttention_59313498357770
// MI455X (gfx1250) — matrix operand order repaired and checked by reading, not compiled, not run on hardware
//
#include <hip/hip_runtime.h>

#define B_N 32
#define S_N 2048
#define D_N 64

typedef __bf16 v16bf __attribute__((ext_vector_type(16)));
typedef float  v8f   __attribute__((ext_vector_type(8)));

#if __has_builtin(__builtin_amdgcn_exp2f)
#define EXP2F(x) __builtin_amdgcn_exp2f(x)
#else
#define EXP2F(x) exp2f(x)
#endif

static __device__ __forceinline__ v8f wmma_bf16(v16bf a, v16bf b, v8f c) {
  // (neg_a, A, neg_b, B, c_mod, C, reuse_a, reuse_b)
  return __builtin_amdgcn_wmma_f32_16x16x32_bf16(false, a, false, b, (short)0, c,
                                                 false, false);
}

// A-layout k index for element e of a 16x32 16-bit A fragment (contiguous, matches B):
// e=0..15 -> k = hi*16+e
static __device__ __forceinline__ int k_of_e(int e, int hi) {
  return hi * 16 + e;
}

// ---------------------------------------------------------------------------
// Swizzle pass: write K and V as ready-to-load WMMA A-fragments (bf16).
// KA[b][skT16 (128)][j (2)][lane (32)][e (16)]  : A tile of K  (M=key, K=d)
//   element = keys[b][ d = j*32 + k_of_e(e,hi) ][ s = skT16*16 + (lane&15) ]
// VA[b][skT32 (64)][t (4)][lane (32)][e (16)]   : A tile of V^T (M=d,  K=key)
//   element = values[b][ s = skT32*32 + k_of_e(e,hi) ][ d = t*16 + (lane&15) ]
// In the main loop each fragment is ONE 32-byte vector load per lane.
// ---------------------------------------------------------------------------
__global__ __launch_bounds__(256)
void swz_k_kernel(const float* __restrict__ K, __bf16* __restrict__ KA) {
  int t = blockIdx.x * 256 + threadIdx.x;
  int e = t & 15, lane = (t >> 4) & 31, j = (t >> 9) & 1;
  int skT = (t >> 10) & 127, b = t >> 17;
  int hi = lane >> 4, n = lane & 15;
  int d = j * 32 + k_of_e(e, hi);
  int s = skT * 16 + n;
  KA[t] = (__bf16)K[((size_t)b * D_N + d) * S_N + s];
}

__global__ __launch_bounds__(256)
void swz_v_kernel(const float* __restrict__ V, __bf16* __restrict__ VA) {
  int t = blockIdx.x * 256 + threadIdx.x;
  int e = t & 15, lane = (t >> 4) & 31, tt = (t >> 9) & 3;
  int skT = (t >> 11) & 63, b = t >> 17;
  int hi = lane >> 4, n = lane & 15;
  int s = skT * 32 + k_of_e(e, hi);
  int d = tt * 16 + n;
  VA[t] = (__bf16)V[((size_t)b * S_N + s) * D_N + d];
}

// ---------------------------------------------------------------------------
// Wave32 flash attention, transposed orientation:
//   GEMM1: S^T = K_tile (A) x Q^T (B)   -> each lane owns one query column
//   GEMM2: O^T = V^T_tile (A) x P^T (B) -> P^T built from registers + 1 xor16
// No LDS tiles, no __syncthreads (EXEC stays all-ones for WMMA).
// ---------------------------------------------------------------------------
__global__ __launch_bounds__(256)
void fattn_kernel(const float* __restrict__ Q, const v16bf* __restrict__ KA,
                  const v16bf* __restrict__ VA, float* __restrict__ O) {
  const int lane = threadIdx.x & 31;
  const int n16 = lane & 15;  // this lane's query column
  const int hi = lane >> 4;
  const int wave = threadIdx.x >> 5;
  const int b = blockIdx.x >> 4;
  const int q0 = (blockIdx.x & 15) * 128 + wave * 16;

  // 1/sqrt(64) * log2(e): folded into Q so softmax runs in exp2 domain.
  const float SCALE = 0.125f * 1.44269504088896340736f;

  // Q^T B-fragments (K-dim=d): lane col = query n16, element e -> d = j*32+hi*16+e
  // => 16 consecutive floats of one Q row: pure vector loads.
  v16bf bQ[2];
  const float* qrow = Q + ((size_t)b * S_N + q0 + n16) * D_N;
#pragma unroll
  for (int j = 0; j < 2; j++) {
    const float* base = qrow + j * 32 + hi * 16;
#pragma unroll
    for (int q4 = 0; q4 < 4; q4++) {
      float4 f = ((const float4*)base)[q4];
      const int e = q4 * 4;
      bQ[j][e + 0] = (__bf16)(f.x * SCALE);
      bQ[j][e + 1] = (__bf16)(f.y * SCALE);
      bQ[j][e + 2] = (__bf16)(f.z * SCALE);
      bQ[j][e + 3] = (__bf16)(f.w * SCALE);
    }
  }

  // O^T accumulators: acc[t][r] -> d = t*16 + hi*8 + r, query = n16
  const v8f vzero = {0.f, 0.f, 0.f, 0.f, 0.f, 0.f, 0.f, 0.f};
  v8f acc[4];
#pragma unroll
  for (int t = 0; t < 4; t++) acc[t] = vzero;
  float m = -1e30f, l = 0.f;  // per-query scalars (lane & lane^16 stay in sync)

  const v16bf* kfrag = KA + (size_t)b * 128 * 2 * 32;
  const v16bf* vfrag = VA + (size_t)b * 64 * 4 * 32;

  for (int it = 0; it < S_N / 32; ++it) {
    __builtin_prefetch(&kfrag[(size_t)((it + 1) * 4) * 32 + lane], 0, 0);
    __builtin_prefetch(&vfrag[(size_t)((it + 1) * 4) * 32 + lane], 0, 0);

    // ---- issue ALL fragment loads up front: V loads are independent of the
    // scores, so they fill the WMMA->VALU hazard slots and cover PV latency.
    v16bf aK0 = kfrag[(size_t)((it * 2 + 0) * 2 + 0) * 32 + lane];
    v16bf aK1 = kfrag[(size_t)((it * 2 + 0) * 2 + 1) * 32 + lane];
    v16bf aK2 = kfrag[(size_t)((it * 2 + 1) * 2 + 0) * 32 + lane];
    v16bf aK3 = kfrag[(size_t)((it * 2 + 1) * 2 + 1) * 32 + lane];
    v16bf aV0 = vfrag[(size_t)(it * 4 + 0) * 32 + lane];
    v16bf aV1 = vfrag[(size_t)(it * 4 + 1) * 32 + lane];
    v16bf aV2 = vfrag[(size_t)(it * 4 + 2) * 32 + lane];
    v16bf aV3 = vfrag[(size_t)(it * 4 + 3) * 32 + lane];

    // ---- S^T tiles: sc[c][r] = score(key = it*32 + c*16 + hi*8 + r, query n16)
    v8f sc[2];
    {
      v8f s0 = wmma_bf16(aK0, bQ[0], vzero);
      s0 = wmma_bf16(aK1, bQ[1], s0);
      v8f s1 = wmma_bf16(aK2, bQ[0], vzero);
      s1 = wmma_bf16(aK3, bQ[1], s1);
      sc[0] = s0;
      sc[1] = s1;
    }

    // ---- online softmax over keys: in-lane tree + ONE xor-16 combine
    float vmax = fmaxf(sc[0][0], sc[1][0]);
#pragma unroll
    for (int r = 1; r < 8; r++) vmax = fmaxf(vmax, fmaxf(sc[0][r], sc[1][r]));
    vmax = fmaxf(vmax, __shfl_xor(vmax, 16, 32));
    float mn = fmaxf(m, vmax);
    float corr = EXP2F(m - mn);
    m = mn;

    float p0[8], p1[8];
    float rs = 0.f;
#pragma unroll
    for (int r = 0; r < 8; r++) {
      p0[r] = EXP2F(sc[0][r] - mn);
      p1[r] = EXP2F(sc[1][r] - mn);
      rs += p0[r] + p1[r];
    }
    rs += __shfl_xor(rs, 16, 32);
    l = l * corr + rs;
#pragma unroll
    for (int t = 0; t < 4; t++) acc[t] = acc[t] * corr;

    // ---- build P^T B-fragment: lane needs keys hi*16+e for its query.
    // Own regs cover e = hi*8..hi*8+7 (block c=hi); partner lane^16 covers the
    // other 8 -> one xor-16 exchange of the complementary block.
    v16bf bP;
#pragma unroll
    for (int r = 0; r < 8; r++) {
      float own = hi ? p1[r] : p0[r];           // my p[c=hi][r]
      float send = hi ? p0[r] : p1[r];          // my p[c=1-hi][r] for partner
      float rec = __shfl_xor(send, 16, 32);     // partner's p[c=my hi][r]
      float lo = hi ? rec : own;                // element e = r
      float hv = hi ? own : rec;                // element e = 8 + r
      bP[r] = (__bf16)lo;
      bP[8 + r] = (__bf16)hv;
    }

    // ---- O^T += V^T x P^T
    acc[0] = wmma_bf16(aV0, bP, acc[0]);
    acc[1] = wmma_bf16(aV1, bP, acc[1]);
    acc[2] = wmma_bf16(aV2, bP, acc[2]);
    acc[3] = wmma_bf16(aV3, bP, acc[3]);
  }

  // ---- epilogue: O^T C-layout -> global row-major, vector stores
  float invl = 1.0f / l;
  float* orow = O + ((size_t)b * S_N + q0 + n16) * D_N + hi * 8;
#pragma unroll
  for (int t = 0; t < 4; t++) {
    float4 o0 = {acc[t][0] * invl, acc[t][1] * invl, acc[t][2] * invl,
                 acc[t][3] * invl};
    float4 o1 = {acc[t][4] * invl, acc[t][5] * invl, acc[t][6] * invl,
                 acc[t][7] * invl};
    *(float4*)(orow + t * 16) = o0;
    *(float4*)(orow + t * 16 + 4) = o1;
  }
}

// ---------------------------------------------------------------------------
extern "C" void kernel_launch(void* const* d_in, const int* in_sizes, int n_in,
                              void* d_out, int out_size, void* d_ws, size_t ws_size,
                              hipStream_t stream) {
  const float* q = (const float*)d_in[0];
  const float* k = (const float*)d_in[1];  // [B, D, S] (pre-transposed K)
  const float* v = (const float*)d_in[2];  // [B, S, D]
  float* out = (float*)d_out;

  __bf16* ka = (__bf16*)d_ws;
  __bf16* va = (__bf16*)((char*)d_ws + (size_t)B_N * D_N * S_N * sizeof(__bf16));

  const int n = B_N * S_N * D_N;  // 4,194,304 fragment elements per tensor
  swz_k_kernel<<<n / 256, 256, 0, stream>>>(k, ka);
  swz_v_kernel<<<n / 256, 256, 0, stream>>>(v, va);

  fattn_kernel<<<B_N * (S_N / 128), 256, 0, stream>>>(
      q, (const v16bf*)ka, (const v16bf*)va, out);
}
